// TDMPSMNISTSNN_15831249453629
// MI455X (gfx1250) — matrix operand order repaired and checked by reading, not compiled, not run on hardware
//
#include <hip/hip_runtime.h>

// ---------------------------------------------------------------------------
// Per-step recurrent SNN with bank-switched membranes.
//  - 16 batch rows per workgroup, 8 waves x 2 N-tiles cover N_HID=256
//  - spk @ Wrec.T via v_wmma_f32_16x16x32_f16 (A = spikes, exact in fp16)
//  - Wrec(fp16) + spk buffer + W2 in LDS; membrane banks in VGPRs
// ---------------------------------------------------------------------------

typedef _Float16 v8h  __attribute__((ext_vector_type(8)));
typedef _Float16 v16h __attribute__((ext_vector_type(16)));
typedef float    v8f  __attribute__((ext_vector_type(8)));

#define T_STEPS   784
#define N_HID     256
#define N_OUT     10
#define ROWS      16          // batch rows per workgroup
#define LDS_WREC  (N_HID * N_HID * 2)           // 131072 B, fp16
#define LDS_SPK   (ROWS * N_HID * 2)            // 8192 B, fp16
#define LDS_W2    (N_OUT * N_HID * 4)           // 10240 B, f32
#define LDS_XBUF  (ROWS * 4)                    // 64 B, f32
#define LDS_TOTAL (LDS_WREC + LDS_SPK + LDS_W2 + LDS_XBUF)

__global__ void __launch_bounds__(256)
snn_scan_kernel(const float* __restrict__ x,
                const float* __restrict__ W1,
                const float* __restrict__ Wrec,
                const float* __restrict__ W2,
                const float* __restrict__ beta1,
                const float* __restrict__ beta2p,
                float* __restrict__ out)
{
    extern __shared__ char smem[];
    _Float16* wrec_lds = (_Float16*)(smem);
    _Float16* spk_lds  = (_Float16*)(smem + LDS_WREC);
    float*    w2_lds   = (float*)   (smem + LDS_WREC + LDS_SPK);
    float*    xbuf     = (float*)   (smem + LDS_WREC + LDS_SPK + LDS_W2);

    const int tid  = threadIdx.x;          // 0..255  (8 waves, wave32)
    const int lane = tid & 31;
    const int wave = tid >> 5;
    const int half = lane >> 4;            // 0: lanes 0-15, 1: lanes 16-31
    const int lm   = lane & 15;
    const int base = blockIdx.x * ROWS;    // first batch row of this WG

    // ---- one-time prep: fp16 Wrec -> LDS, W2 -> LDS, zero spike buffer ----
    for (int i = tid; i < N_HID * N_HID; i += 256)
        wrec_lds[i] = (_Float16)Wrec[i];
    for (int i = tid; i < N_OUT * N_HID; i += 256)
        w2_lds[i] = W2[i];
    for (int i = tid; i < ROWS * N_HID; i += 256)
        spk_lds[i] = (_Float16)0.0f;

    // per-lane constants for the 2 N-tiles this wave owns
    int   Ng[2];
    float w1v[2], b1v[2];
#pragma unroll
    for (int tl = 0; tl < 2; ++tl) {
        Ng[tl]  = (wave * 2 + tl) * 16 + lm;   // hidden index (column N)
        w1v[tl] = W1[Ng[tl]];                  // W1 is (256,1)
        b1v[tl] = beta1[Ng[tl]];
    }
    const float beta2 = beta2p[0];

    // membrane banks, D-layout: element r holds row (r + 8*half)
    v8f vbank[2][4];
#pragma unroll
    for (int tl = 0; tl < 2; ++tl)
#pragma unroll
        for (int b = 0; b < 4; ++b)
#pragma unroll
            for (int r = 0; r < 8; ++r)
                vbank[tl][b][r] = 0.0f;

    // readout state: thread (tid<160) owns (row om, out oo)
    const int om = tid & 15;
    const int oo = tid >> 4;
    float v2 = 0.0f, osum = 0.0f;

    __syncthreads();

    for (int tb = 0; tb < T_STEPS; tb += 4) {
#pragma unroll
        for (int q = 0; q < 4; ++q) {          // q == bank (t % 4), compile-time
            const int t = tb + q;

            // stage x[:, t] for the 16 rows
            if (tid < ROWS) xbuf[tid] = x[(base + tid) * T_STEPS + t];
            __syncthreads();   // prev-step spikes + xbuf visible

            float xr[8];
#pragma unroll
            for (int r = 0; r < 8; ++r) xr[r] = xbuf[r + half * 8];

            // ---- I_rec = spk_prev @ Wrec.T via WMMA ----
            v8f c[2];
#pragma unroll
            for (int tl = 0; tl < 2; ++tl)
#pragma unroll
                for (int r = 0; r < 8; ++r) c[tl][r] = 0.0f;

            const _Float16* arow = spk_lds + lm * N_HID;   // A row M = lane%16
#pragma unroll
            for (int k = 0; k < 8; ++k) {                  // K chunks of 32
                v8h alo = *(const v8h*)(arow + k * 32 + half * 16);
                v8h ahi = *(const v8h*)(arow + k * 32 + half * 16 + 8);
                v16h a = __builtin_shufflevector(alo, ahi,
                         0,1,2,3,4,5,6,7,8,9,10,11,12,13,14,15);
#pragma unroll
                for (int tl = 0; tl < 2; ++tl) {
                    const _Float16* bp =
                        wrec_lds + Ng[tl] * N_HID + k * 32 + half * 16;
                    v8h blo = *(const v8h*)(bp);
                    v8h bhi = *(const v8h*)(bp + 8);
                    v16h bm = __builtin_shufflevector(blo, bhi,
                              0,1,2,3,4,5,6,7,8,9,10,11,12,13,14,15);
                    c[tl] = __builtin_amdgcn_wmma_f32_16x16x32_f16(
                        false, a, false, bm, (short)0, c[tl], false, false);
                }
            }
            __syncthreads();   // all reads of spk_lds done before overwrite

            // ---- membrane update, threshold, write new spikes ----
#pragma unroll
            for (int tl = 0; tl < 2; ++tl) {
#pragma unroll
                for (int r = 0; r < 8; ++r) {
                    float v = b1v[tl] * vbank[tl][q][r] + c[tl][r]
                            + xr[r] * w1v[tl];
                    float s = (v >= 1.0f) ? 1.0f : 0.0f;
                    v -= s;                       // soft reset by threshold
                    vbank[tl][q][r] = v;
                    spk_lds[(r + half * 8) * N_HID + Ng[tl]] = (_Float16)s;
                }
            }
            __syncthreads();   // new spikes visible

            // ---- readout: I2 = spk @ W2.T, leaky accumulate ----
            if (tid < ROWS * N_OUT) {
                const _Float16* sp = spk_lds + om * N_HID;
                const float*    wp = w2_lds + oo * N_HID;
                float acc = 0.0f;
#pragma unroll 8
                for (int h = 0; h < N_HID; ++h)
                    acc += (float)sp[h] * wp[h];
                v2 = beta2 * v2 + (1.0f - beta2) * acc;
                osum += v2;
            }
        }
    }

    if (tid < ROWS * N_OUT)
        out[(base + om) * N_OUT + oo] = osum * (1.0f / (float)T_STEPS);
}

extern "C" void kernel_launch(void* const* d_in, const int* in_sizes, int n_in,
                              void* d_out, int out_size, void* d_ws, size_t ws_size,
                              hipStream_t stream) {
    const float* x     = (const float*)d_in[0];   // (1024, 784, 1)
    const float* W1    = (const float*)d_in[1];   // (256, 1)
    const float* Wrec  = (const float*)d_in[2];   // (256, 256)
    const float* W2    = (const float*)d_in[3];   // (10, 256)
    const float* beta1 = (const float*)d_in[4];   // (256,)
    const float* beta2 = (const float*)d_in[5];   // scalar
    float* out = (float*)d_out;                   // (1024, 10)

    // allow >64KB dynamic LDS (320KB per WGP on gfx1250)
    (void)hipFuncSetAttribute((const void*)snn_scan_kernel,
                              hipFuncAttributeMaxDynamicSharedMemorySize,
                              LDS_TOTAL);

    dim3 grid(1024 / ROWS);   // 64 workgroups
    dim3 block(256);          // 8 wave32 waves
    snn_scan_kernel<<<grid, block, LDS_TOTAL, stream>>>(
        x, W1, Wrec, W2, beta1, beta2, out);
    (void)d_ws; (void)ws_size; (void)in_sizes; (void)n_in; (void)out_size;
}
